// ActuatorNet_58437325029824
// MI455X (gfx1250) — matrix operand order repaired and checked by reading, not compiled, not run on hardware
//
#include <hip/hip_runtime.h>
#include <hip/hip_bf16.h>

typedef __attribute__((ext_vector_type(16))) _Float16 v16h;
typedef __attribute__((ext_vector_type(2)))  __fp16   v2fp;   // cvt_pkrtz result type
typedef __attribute__((ext_vector_type(8)))  float    v8f;

#define WMMA_F16(A, B, C) \
  __builtin_amdgcn_wmma_f32_16x16x32_f16(false, (A), false, (B), (short)0, (C), false, false)

#if __has_builtin(__builtin_amdgcn_tanhf)
  #define TANHF(x) __builtin_amdgcn_tanhf(x)
  #define HAVE_HW_TANH 1
#elif __has_builtin(__builtin_amdgcn_tanh_f32)
  #define TANHF(x) __builtin_amdgcn_tanh_f32(x)
  #define HAVE_HW_TANH 1
#else
  #define HAVE_HW_TANH 0
#endif

union V16 {
  v16h h;
  uint4 u[2];
  _Float16 e[16];
};

__device__ __forceinline__ float fast_sigmoid(float x) {
#if HAVE_HW_TANH
  return fmaf(0.5f, TANHF(0.5f * x), 0.5f);       // v_mul, v_tanh, v_fma
#else
  float e = __expf(-x);
  return __builtin_amdgcn_rcpf(1.0f + e);
#endif
}

__device__ __forceinline__ float fast_tanh(float x) {
#if HAVE_HW_TANH
  return TANHF(x);                                 // single v_tanh_f32
#else
  float e = __expf(2.0f * x);
  return 1.0f - 2.0f * __builtin_amdgcn_rcpf(e + 1.0f);
#endif
}

__device__ __forceinline__ uint4 pack8(v8f v) {
  union { uint4 u; v2fp h[4]; } P;
  P.h[0] = __builtin_amdgcn_cvt_pkrtz(v[0], v[1]);  // v_cvt_pk_rtz_f16_f32
  P.h[1] = __builtin_amdgcn_cvt_pkrtz(v[2], v[3]);
  P.h[2] = __builtin_amdgcn_cvt_pkrtz(v[4], v[5]);
  P.h[3] = __builtin_amdgcn_cvt_pkrtz(v[6], v[7]);
  return P.u;
}

// ---------------------------------------------------------------------------
// Prep: padded f16 weight images in workspace.
//   Per layer: [96 rows][32 cols] f16; rows = 3 gate blocks (r,z,n) of 32
//   (28 real + 4 zero pad); col 28 = bias; cols beyond real K are zero.
// ---------------------------------------------------------------------------
__global__ void gru_prep_kernel(const float* __restrict__ Wih0,
                                const float* __restrict__ WihR,
                                const float* __restrict__ Whh,
                                const float* __restrict__ bih,
                                const float* __restrict__ bhh,
                                _Float16* __restrict__ wihP,
                                _Float16* __restrict__ whhP) {
  int idx = blockIdx.x * 256 + threadIdx.x;
  if (idx >= 4 * 96 * 32) return;
  int k   = idx & 31;
  int row = (idx >> 5) % 96;
  int l   = idx / (96 * 32);
  int g   = row >> 5;   // gate block: 0=r 1=z 2=n
  int i   = row & 31;   // row within block
  float vih = 0.0f, vhh = 0.0f;
  if (i < 28) {
    int grow = g * 28 + i;              // original gate row 0..83
    if (k == 28) {
      vih = bih[l * 84 + grow];
      vhh = bhh[l * 84 + grow];
    } else if (k < 28) {
      vhh = Whh[(l * 84 + grow) * 28 + k];
      if (l == 0) {
        if (k < 2) vih = Wih0[grow * 2 + k];
      } else {
        vih = WihR[((l - 1) * 84 + grow) * 28 + k];
      }
    }
  }
  wihP[idx] = (_Float16)vih;
  whhP[idx] = (_Float16)vhh;
}

// ---------------------------------------------------------------------------
// Main GRU kernel. Block = 64 threads = 2 waves; each wave owns 16 batch
// rows through 4 layers x 28 steps. History slab (29 slots):
//   slot 0      : initial state (zeros, 1.0 at K=28) -- never overwritten,
//                 doubles as h0 for every layer.
//   slots 1..28 : layer input at step t (pre-filled with padded x), each is
//                 overwritten in place with the layer's output h_t.
// Per (layer, step): 12 v_wmma_f32_16x16x32_f16, zero branches.
// ---------------------------------------------------------------------------
__global__ __launch_bounds__(64) void gru_main_kernel(
    const float* __restrict__ x,        // [B][28][2]
    const _Float16* __restrict__ wihP,  // [4][96][32]
    const _Float16* __restrict__ whhP,  // [4][96][32]
    const float* __restrict__ fcw,      // [28]
    const float* __restrict__ fcb,      // [1]
    float* __restrict__ out) {          // [B]
  __shared__ _Float16 hist[2][29][16][32];

  const int lane = threadIdx.x & 31;
  const int w    = threadIdx.x >> 5;
  const int col  = lane & 15;   // batch column / A-tile row offset
  const int s    = lane >> 4;   // half-wave group
  const long batch0 = (long)blockIdx.x * 32 + (long)w * 16;

  _Float16 (*H)[16][32] = hist[w];

  // ---- preamble: initial state slot + x pre-fill (padded B-operand form) --
  {
    const uint4 zz   = make_uint4(0u, 0u, 0u, 0u);
    const uint4 bias = make_uint4(0u, 0u, 0x00003C00u, 0u);  // half 28 = 1.0

    // slot 0: initial hidden state
    if (s == 0) {
      *(uint4*)&H[0][col][0] = zz;
      *(uint4*)&H[0][col][8] = zz;
    } else {
      *(uint4*)&H[0][col][16] = zz;
      *(uint4*)&H[0][col][24] = bias;
    }

    // slots 1..28: x_t at K=0,1 ; zeros ; 1.0 at K=28
    const float* xrow = x + (batch0 + col) * 56;
#pragma unroll
    for (int tp = 0; tp < 14; ++tp) {
      if (s == 0) {
        float4 xv = *(const float4*)(xrow + tp * 4);     // t=2tp and t=2tp+1
        union { uint4 u; v2fp h[4]; } P0, P1;
        P0.u = zz; P1.u = zz;
        P0.h[0] = __builtin_amdgcn_cvt_pkrtz(xv.x, xv.y);
        P1.h[0] = __builtin_amdgcn_cvt_pkrtz(xv.z, xv.w);
        *(uint4*)&H[1 + 2 * tp][col][0] = P0.u;
        *(uint4*)&H[1 + 2 * tp][col][8] = zz;
        *(uint4*)&H[2 + 2 * tp][col][0] = P1.u;
        *(uint4*)&H[2 + 2 * tp][col][8] = zz;
      } else {
        *(uint4*)&H[1 + 2 * tp][col][16] = zz;
        *(uint4*)&H[1 + 2 * tp][col][24] = bias;
        *(uint4*)&H[2 + 2 * tp][col][16] = zz;
        *(uint4*)&H[2 + 2 * tp][col][24] = bias;
      }
    }
  }

#pragma unroll 1
  for (int l = 0; l < 4; ++l) {
    // ---- resident A-tiles for this layer (96 VGPRs) ----
    const _Float16* wih = wihP + l * 96 * 32;
    const _Float16* whh = whhP + l * 96 * 32;
    V16 Aih[6], Ahh[6];
#pragma unroll
    for (int tt = 0; tt < 6; ++tt) {
      const int row = tt * 16 + col;
      const int kb  = 8 * s;
      Aih[tt].u[0] = *(const uint4*)(wih + row * 32 + kb);
      Aih[tt].u[1] = *(const uint4*)(wih + row * 32 + kb + 16);
      Ahh[tt].u[0] = *(const uint4*)(whh + row * 32 + kb);
      Ahh[tt].u[1] = *(const uint4*)(whh + row * 32 + kb + 16);
    }

    v8f hold0 = {0, 0, 0, 0, 0, 0, 0, 0};  // h_old, D-layout, h 0..15
    v8f hold1 = {0, 0, 0, 0, 0, 0, 0, 0};  // h_old, D-layout, h 16..31

#pragma unroll 1
    for (int t = 0; t < 28; ++t) {
      // B operand: layer input at step t (slot t+1)
      V16 xB;
      xB.u[0] = *(const uint4*)&H[t + 1][col][8 * s];
      xB.u[1] = *(const uint4*)&H[t + 1][col][8 * s + 16];
      // B operand: h_{t-1} (slot t; slot 0 = initial state)
      V16 hB;
      hB.u[0] = *(const uint4*)&H[t][col][8 * s];
      hB.u[1] = *(const uint4*)&H[t][col][8 * s + 16];

      const v8f zero = {0, 0, 0, 0, 0, 0, 0, 0};

      // 12 WMMAs: r/z chained (gi then +gh), n-gate gi/gh kept separate
      v8f aR0 = WMMA_F16(Aih[0].h, xB.h, zero);
      v8f aR1 = WMMA_F16(Aih[1].h, xB.h, zero);
      v8f aZ0 = WMMA_F16(Aih[2].h, xB.h, zero);
      v8f aZ1 = WMMA_F16(Aih[3].h, xB.h, zero);
      v8f nI0 = WMMA_F16(Aih[4].h, xB.h, zero);
      v8f nI1 = WMMA_F16(Aih[5].h, xB.h, zero);
      v8f nH0 = WMMA_F16(Ahh[4].h, hB.h, zero);
      v8f nH1 = WMMA_F16(Ahh[5].h, hB.h, zero);
      aR0 = WMMA_F16(Ahh[0].h, hB.h, aR0);
      aR1 = WMMA_F16(Ahh[1].h, hB.h, aR1);
      aZ0 = WMMA_F16(Ahh[2].h, hB.h, aZ0);
      aZ1 = WMMA_F16(Ahh[3].h, hB.h, aZ1);

      // gate math (v_tanh_f32 path when available; co-executes with XDL)
      v8f hn0, hn1;
#pragma unroll
      for (int v = 0; v < 8; ++v) {
        float r0 = fast_sigmoid(aR0[v]);
        float z0 = fast_sigmoid(aZ0[v]);
        float n0 = fast_tanh(nI0[v] + r0 * nH0[v]);
        hn0[v] = (1.0f - z0) * n0 + z0 * hold0[v];
        float r1 = fast_sigmoid(aR1[v]);
        float z1 = fast_sigmoid(aZ1[v]);
        float n1 = fast_tanh(nI1[v] + r1 * nH1[v]);
        hn1[v] = (1.0f - z1) * n1 + z1 * hold1[v];
      }
      hold0 = hn0;
      hold1 = hn1;

      // store h_t into slot t+1 (D-layout rows are h-contiguous per lane)
      *(uint4*)&H[t + 1][col][8 * s] = pack8(hn0);     // h = 8s+v
      v8f t1 = hn1;                                    // h = 16+8s+v
      if (s == 1) {                                    // force K=28 bias, pad
        t1[4] = 1.0f;
        t1[5] = 0.0f; t1[6] = 0.0f; t1[7] = 0.0f;
      }
      *(uint4*)&H[t + 1][col][16 + 8 * s] = pack8(t1);
    }
  }

  // ---- FC head: out[b] = fc_w . h_last[b] + fc_b ----
  if (s == 0) {
    float acc = fcb[0];
#pragma unroll
    for (int i = 0; i < 28; ++i) acc += fcw[i] * (float)H[28][col][i];
    out[batch0 + col] = acc;
  }
}

extern "C" void kernel_launch(void* const* d_in, const int* in_sizes, int n_in,
                              void* d_out, int out_size, void* d_ws, size_t ws_size,
                              hipStream_t stream) {
  const float* x    = (const float*)d_in[0];  // [B,28,2]
  const float* Wih0 = (const float*)d_in[1];  // [84,2]
  const float* WihR = (const float*)d_in[2];  // [3,84,28]
  const float* Whh  = (const float*)d_in[3];  // [4,84,28]
  const float* bih  = (const float*)d_in[4];  // [4,84]
  const float* bhh  = (const float*)d_in[5];  // [4,84]
  const float* fcw  = (const float*)d_in[6];  // [1,28]
  const float* fcb  = (const float*)d_in[7];  // [1]
  float* out = (float*)d_out;

  _Float16* wihP = (_Float16*)d_ws;
  _Float16* whhP = wihP + 4 * 96 * 32;

  gru_prep_kernel<<<(4 * 96 * 32 + 255) / 256, 256, 0, stream>>>(
      Wih0, WihR, Whh, bih, bhh, wihP, whhP);

  const int B = 262144;
  gru_main_kernel<<<B / 32, 64, 0, stream>>>(x, wihP, whhP, fcw, fcb, out);
}
